// MultiModalGraphSAGE_65584150610482
// MI455X (gfx1250) — matrix operand order repaired and checked by reading, not compiled, not run on hardware
//
#include <hip/hip_runtime.h>
#include <hip/hip_bf16.h>

#define N_NODES 100000
#define N_EDGES 800000
#define ROW_TILES (N_NODES / 16)   // 6250 exactly -> no row tail, EXEC always full

typedef __bf16 bf16_t;
typedef __attribute__((ext_vector_type(16))) __bf16 v16bf;
typedef __attribute__((ext_vector_type(8)))  float  v8f;

__device__ __forceinline__ v16bf frag16(uint4 lo, uint4 hi) {
    union { uint4 u[2]; v16bf b; } t;
    t.u[0] = lo; t.u[1] = hi;
    return t.b;
}

// ---------------------------------------------------------------------------
// Weight pre-pack: f32 row-major W[K][N] -> bf16 packed in WMMA B-fragment
// order.  Packed index = ((kt*N + n)*32 + half*16 + e), element = W[kt*32 +
// half*16 + e][n].  Lane (n = lane&15, half = lane>>4) then reads its 16-elem
// fragment as two contiguous b128 loads.
// ---------------------------------------------------------------------------
__global__ void pack_w(const float* __restrict__ W, bf16_t* __restrict__ P,
                       int K, int Nc) {
    int i = blockIdx.x * blockDim.x + threadIdx.x;
    if (i >= K * Nc) return;
    int e    = i & 15;
    int half = (i >> 4) & 1;
    int n    = (i >> 5) % Nc;
    int kt   = (i >> 5) / Nc;
    int k    = kt * 32 + half * 16 + e;
    P[i] = (bf16_t)W[(size_t)k * Nc + n];
}

// concat(structural[128], multimodal[384]) -> bf16 x[N,512]
__global__ void concat_to_bf(const float* __restrict__ s,
                             const float* __restrict__ m,
                             bf16_t* __restrict__ x) {
    long long i = (long long)blockIdx.x * blockDim.x + threadIdx.x;
    if (i >= (long long)N_NODES * 512) return;
    int c = (int)(i & 511);
    long long row = i >> 9;
    float v = (c < 128) ? s[row * 128 + c] : m[row * 384 + (c - 128)];
    x[i] = (bf16_t)v;
}

__global__ void zero_f32(float* __restrict__ p, long long n) {
    long long i = (long long)blockIdx.x * blockDim.x + threadIdx.x;
    if (i < n) p[i] = 0.0f;
}

__global__ void degree_k(const int* __restrict__ dst, float* __restrict__ deg) {
    int e = blockIdx.x * blockDim.x + threadIdx.x;
    if (e < N_EDGES) atomicAdd(&deg[dst[e]], 1.0f);
}

// Scatter-sum: for each (edge, dword-of-2-bf16), gather h[src] and atomically
// accumulate fp32 into sum[dst].  This is the HBM-bound part of the workload.
__global__ void agg_edges(const int* __restrict__ src, const int* __restrict__ dst,
                          const bf16_t* __restrict__ h, int ldh, int D,
                          float* __restrict__ sum) {
    long long idx = (long long)blockIdx.x * blockDim.x + threadIdx.x;
    int halfD = D >> 1;
    long long total = (long long)N_EDGES * halfD;
    if (idx >= total) return;
    int e  = (int)(idx / halfD);
    int d2 = (int)(idx % halfD);
    int s = src[e];
    int d = dst[e];
    const bf16_t* p = h + (long long)s * ldh + 2 * d2;
    float f0 = (float)p[0];
    float f1 = (float)p[1];
    float* q = sum + (long long)d * D + 2 * d2;
    atomicAdd(q,     f0);
    atomicAdd(q + 1, f1);
}

// mean = sum / max(deg,1), store bf16
__global__ void mean_to_bf(const float* __restrict__ sum, const float* __restrict__ deg,
                           bf16_t* __restrict__ o, int D, long long total) {
    long long i = (long long)blockIdx.x * blockDim.x + threadIdx.x;
    if (i >= total) return;
    long long row = i / D;
    o[i] = (bf16_t)(sum[i] / fmaxf(deg[row], 1.0f));
}

// ---------------------------------------------------------------------------
// Generic WMMA GEMM: D = act(bn(A*B [+ A2*B2] + bias)).
// One wave (32 threads) computes a 16 x (16*NB) output strip.  Per K-step,
// ALL fragments (A [+A2] and the NB B [+B2] fragments) are loaded into
// distinct registers before any WMMA issues, so the loads form one long
// clause and the v_wmma chain drains them with partial s_wait_loadcnt --
// WMMA nb overlaps the still-outstanding loads for nb+1..
// K multiple of 32; rows and cols tile exactly.
// ---------------------------------------------------------------------------
template<int NB>
__global__ __launch_bounds__(32)
void gemm_wmma(const bf16_t* __restrict__ A, int lda, const bf16_t* __restrict__ Bp,
               const bf16_t* __restrict__ A2, int lda2, const bf16_t* __restrict__ Bp2,
               const float* __restrict__ bias,
               const float* __restrict__ bng, const float* __restrict__ bnb,
               const float* __restrict__ bnm, const float* __restrict__ bnv,
               bf16_t* __restrict__ outBf, int ldo, int col_off,
               float* __restrict__ outF,
               int K, int Nc, int relu) {
    const int lane = threadIdx.x;
    const int l16  = lane & 15;
    const int half = lane >> 4;
    const long long m0 = (long long)blockIdx.x * 16;
    const int n0 = blockIdx.y * (16 * NB);

    v8f acc[NB];
#pragma unroll
    for (int nb = 0; nb < NB; ++nb)
        acc[nb] = (v8f){0.f, 0.f, 0.f, 0.f, 0.f, 0.f, 0.f, 0.f};

    const int ksteps = K >> 5;
    const bf16_t* arow  = A  + (m0 + l16) * (long long)lda;
    const bf16_t* arow2 = A2 ? (A2 + (m0 + l16) * (long long)lda2) : nullptr;
    // base fragment pointer for this lane; sub-tile nb sits at +nb*16*32 elems
    const bf16_t* bfrag  = Bp  + ((long long)(n0 + l16) * 32) + half * 16;
    const bf16_t* bfrag2 = Bp2 ? (Bp2 + ((long long)(n0 + l16) * 32) + half * 16) : nullptr;
    const long long bstep = (long long)Nc * 32;   // fragment stride per K-tile

    for (int kt = 0; kt < ksteps; ++kt) {
        if (kt + 1 < ksteps) {
            __builtin_prefetch((const void*)(arow + (kt + 1) * 32), 0, 1);
        }
        // ---- issue ALL loads for this K-step first (distinct registers) ----
        // A fragment: lanes 0-15 hold K[0..15]; lanes 16-31 K[16..31]
        uint4 alo = *(const uint4*)(arow + kt * 32 + half * 16);
        uint4 ahi = *(const uint4*)(arow + kt * 32 + half * 16 + 8);
        const bf16_t* bk = bfrag + (long long)kt * bstep;
        uint4 bl[NB], bh[NB];
#pragma unroll
        for (int nb = 0; nb < NB; ++nb) {
            bl[nb] = *(const uint4*)(bk + nb * (16 * 32));
            bh[nb] = *(const uint4*)(bk + nb * (16 * 32) + 8);
        }
        uint4 a2lo, a2hi, b2l[NB], b2h[NB];
        if (A2) {  // uniform scalar branch: EXEC unchanged
            a2lo = *(const uint4*)(arow2 + kt * 32 + half * 16);
            a2hi = *(const uint4*)(arow2 + kt * 32 + half * 16 + 8);
            const bf16_t* b2k = bfrag2 + (long long)kt * bstep;
#pragma unroll
            for (int nb = 0; nb < NB; ++nb) {
                b2l[nb] = *(const uint4*)(b2k + nb * (16 * 32));
                b2h[nb] = *(const uint4*)(b2k + nb * (16 * 32) + 8);
            }
        }
        // ---- then the WMMA chain: drains loads with partial waits ----
        v16bf afrag = frag16(alo, ahi);
#pragma unroll
        for (int nb = 0; nb < NB; ++nb) {
            acc[nb] = __builtin_amdgcn_wmma_f32_16x16x32_bf16(
                          false, afrag, false, frag16(bl[nb], bh[nb]),
                          (short)0, acc[nb], false, false);
        }
        if (A2) {
            v16bf a2frag = frag16(a2lo, a2hi);
#pragma unroll
            for (int nb = 0; nb < NB; ++nb) {
                acc[nb] = __builtin_amdgcn_wmma_f32_16x16x32_bf16(
                              false, a2frag, false, frag16(b2l[nb], b2h[nb]),
                              (short)0, acc[nb], false, false);
            }
        }
    }

    // Epilogue.  C/D layout: lane holds (m = r + 8*half, n = ntile + (lane&15))
#pragma unroll
    for (int nb = 0; nb < NB; ++nb) {
        const int n = n0 + nb * 16 + l16;
        float bia = bias ? bias[n] : 0.0f;
        float g = 1.0f, bb = 0.0f, mu = 0.0f, inv = 1.0f;
        if (bng) {
            g = bng[n]; bb = bnb[n]; mu = bnm[n];
            inv = rsqrtf(bnv[n] + 1e-5f);
        }
#pragma unroll
        for (int r = 0; r < 8; ++r) {
            long long m = m0 + half * 8 + r;
            float val = acc[nb][r] + bia;
            if (bng)  val = (val - mu) * (g * inv) + bb;
            if (relu) val = fmaxf(val, 0.0f);
            if (outBf) outBf[m * ldo + col_off + n] = (bf16_t)val;
            if (outF)  outF[m * Nc + n] = val;
        }
    }
}

// ---------------------------------------------------------------------------
// Host orchestration
// ---------------------------------------------------------------------------
static inline long long cdiv(long long a, long long b) { return (a + b - 1) / b; }

extern "C" void kernel_launch(void* const* d_in, const int* in_sizes, int n_in,
                              void* d_out, int out_size, void* d_ws, size_t ws_size,
                              hipStream_t stream) {
    const float* sfeat = (const float*)d_in[0];
    const float* mfeat = (const float*)d_in[1];
    const int*   src   = (const int*)d_in[2];
    const int*   dst   = (const int*)d_in[3];
    const float* enc_w = (const float*)d_in[4];
    const float* enc_b = (const float*)d_in[5];
    const float* s0_ws = (const float*)d_in[6];
    const float* s0_wn = (const float*)d_in[7];
    const float* s0_b  = (const float*)d_in[8];
    const float* bn0_g = (const float*)d_in[9];
    const float* bn0_b = (const float*)d_in[10];
    const float* bn0_m = (const float*)d_in[11];
    const float* bn0_v = (const float*)d_in[12];
    const float* s1_ws = (const float*)d_in[13];
    const float* s1_wn = (const float*)d_in[14];
    const float* s1_b  = (const float*)d_in[15];
    const float* bn1_g = (const float*)d_in[16];
    const float* bn1_b = (const float*)d_in[17];
    const float* bn1_m = (const float*)d_in[18];
    const float* bn1_v = (const float*)d_in[19];
    const float* rel_w = (const float*)d_in[20];
    const float* rel_b = (const float*)d_in[21];
    const float* c_w1  = (const float*)d_in[22];
    const float* c_b1  = (const float*)d_in[23];
    const float* c_w2  = (const float*)d_in[24];
    const float* c_b2  = (const float*)d_in[25];
    float* out = (float*)d_out;

    // Workspace carve-up (256B aligned).  xbf [N,512] bf16 is reused as the
    // fp32 aggregation accumulator [N,256] (identical 102.4MB footprint).
    char* ws = (char*)d_ws;
    size_t off = 0;
    auto take = [&](size_t bytes) -> char* {
        char* p = ws + off;
        off += (bytes + 255) & ~(size_t)255;
        return p;
    };
    bf16_t* xbf    = (bf16_t*)take((size_t)N_NODES * 512 * 2);
    float*  aggsum = (float*)xbf;                               // alias (post-encode)
    bf16_t* hfbf   = (bf16_t*)take((size_t)N_NODES * 256 * 2);  // [h0 | h2] concat
    bf16_t* h1bf   = (bf16_t*)take((size_t)N_NODES * 256 * 2);  // sage0 out; reused as c1
    bf16_t* mbf    = (bf16_t*)take((size_t)N_NODES * 256 * 2);  // m0/m1; reused as hr
    float*  deg    = (float*)take((size_t)N_NODES * 4);
    bf16_t* encP   = (bf16_t*)take((size_t)512 * 128 * 2);
    bf16_t* s0wsP  = (bf16_t*)take((size_t)128 * 256 * 2);
    bf16_t* s0wnP  = (bf16_t*)take((size_t)128 * 256 * 2);
    bf16_t* s1wsP  = (bf16_t*)take((size_t)256 * 128 * 2);
    bf16_t* s1wnP  = (bf16_t*)take((size_t)256 * 128 * 2);
    bf16_t* relP   = (bf16_t*)take((size_t)256 * 128 * 2);
    bf16_t* w1P    = (bf16_t*)take((size_t)128 * 64 * 2);
    bf16_t* w2P    = (bf16_t*)take((size_t)64 * 32 * 2);

    auto pack = [&](const float* W, bf16_t* P, int K, int Nc) {
        int n = K * Nc;
        pack_w<<<dim3((n + 255) / 256), dim3(256), 0, stream>>>(W, P, K, Nc);
    };
    // NB=4 strip (cols multiple of 64)
    auto gemm4 = [&](const bf16_t* A, int lda, const bf16_t* Bp,
                     const bf16_t* A2, int lda2, const bf16_t* Bp2,
                     const float* bias, const float* g, const float* b,
                     const float* m, const float* v,
                     bf16_t* oBf, int ldo, int coff, float* oF,
                     int K, int Nc, int relu) {
        gemm_wmma<4><<<dim3(ROW_TILES, Nc / 64), dim3(32), 0, stream>>>(
            A, lda, Bp, A2, lda2, Bp2, bias, g, b, m, v,
            oBf, ldo, coff, oF, K, Nc, relu);
    };
    // NB=2 strip (32-wide classifier tail)
    auto gemm2 = [&](const bf16_t* A, int lda, const bf16_t* Bp,
                     const float* bias, bf16_t* oBf, int ldo, float* oF,
                     int K, int Nc, int relu) {
        gemm_wmma<2><<<dim3(ROW_TILES, Nc / 32), dim3(32), 0, stream>>>(
            A, lda, Bp, nullptr, 0, nullptr, bias,
            nullptr, nullptr, nullptr, nullptr,
            oBf, ldo, 0, oF, K, Nc, relu);
    };

    // 0) pack all weights to WMMA B-fragment order (bf16)
    pack(enc_w, encP, 512, 128);
    pack(s0_ws, s0wsP, 128, 256);
    pack(s0_wn, s0wnP, 128, 256);
    pack(s1_ws, s1wsP, 256, 128);
    pack(s1_wn, s1wnP, 256, 128);
    pack(rel_w, relP, 256, 128);
    pack(c_w1,  w1P, 128, 64);
    pack(c_w2,  w2P, 64, 32);

    // 1) x = concat(structural, multimodal) -> bf16
    {
        long long n = (long long)N_NODES * 512;
        concat_to_bf<<<dim3((unsigned)cdiv(n, 256)), dim3(256), 0, stream>>>(sfeat, mfeat, xbf);
    }

    // 2) in-degree (fp32)
    zero_f32<<<dim3((unsigned)cdiv(N_NODES, 256)), dim3(256), 0, stream>>>(deg, N_NODES);
    degree_k<<<dim3((N_EDGES + 255) / 256), dim3(256), 0, stream>>>(dst, deg);

    // 3) h0 = relu(x @ enc_w + enc_b) -> left half of concat buffer [N,256]
    gemm4(xbf, 512, encP, nullptr, 0, nullptr, enc_b,
          nullptr, nullptr, nullptr, nullptr,
          hfbf, 256, 0, nullptr, 512, 128, 1);

    // 4) m0 = mean_agg(h0)
    {
        long long n = (long long)N_NODES * 128;
        zero_f32<<<dim3((unsigned)cdiv(n, 256)), dim3(256), 0, stream>>>(aggsum, n);
        long long t = (long long)N_EDGES * 64;
        agg_edges<<<dim3((unsigned)cdiv(t, 256)), dim3(256), 0, stream>>>(src, dst, hfbf, 256, 128, aggsum);
        mean_to_bf<<<dim3((unsigned)cdiv(n, 256)), dim3(256), 0, stream>>>(aggsum, deg, mbf, 128, n);
    }

    // 5) h1 = relu(bn0(h0@ws + m0@wn + b))  [N,256]
    gemm4(hfbf, 256, s0wsP, mbf, 128, s0wnP, s0_b,
          bn0_g, bn0_b, bn0_m, bn0_v,
          h1bf, 256, 0, nullptr, 128, 256, 1);

    // 6) m1 = mean_agg(h1)
    {
        long long n = (long long)N_NODES * 256;
        zero_f32<<<dim3((unsigned)cdiv(n, 256)), dim3(256), 0, stream>>>(aggsum, n);
        long long t = (long long)N_EDGES * 128;
        agg_edges<<<dim3((unsigned)cdiv(t, 256)), dim3(256), 0, stream>>>(src, dst, h1bf, 256, 256, aggsum);
        mean_to_bf<<<dim3((unsigned)cdiv(n, 256)), dim3(256), 0, stream>>>(aggsum, deg, mbf, 256, n);
    }

    // 7) h2 = relu(bn1(h1@ws + m1@wn + b)) -> right half of concat buffer
    gemm4(h1bf, 256, s1wsP, mbf, 256, s1wnP, s1_b,
          bn1_g, bn1_b, bn1_m, bn1_v,
          hfbf, 256, 128, nullptr, 256, 128, 1);

    // 8) hr = relu(h_final @ rel_w + rel_b)   (hr aliases mbf, now free)
    bf16_t* hr = mbf;
    gemm4(hfbf, 256, relP, nullptr, 0, nullptr, rel_b,
          nullptr, nullptr, nullptr, nullptr,
          hr, 128, 0, nullptr, 256, 128, 1);

    // 9) c1 = relu(hr @ cls_w1 + b1)          (c1 aliases h1bf, now free)
    bf16_t* c1 = h1bf;
    gemm4(hr, 128, w1P, nullptr, 0, nullptr, c_b1,
          nullptr, nullptr, nullptr, nullptr,
          c1, 64, 0, nullptr, 128, 64, 1);

    // 10) out = c1 @ cls_w2 + b2  (fp32 -> d_out), 32 cols -> NB=2
    gemm2(c1, 64, w2P, c_b2, nullptr, 0, out, 64, 32, 0);
}
